// neuralFSI_59777354825820
// MI455X (gfx1250) — matrix operand order repaired and checked by reading, not compiled, not run on hardware
//
#include <hip/hip_runtime.h>
#include <hip/hip_bf16.h>
#include <math.h>
#include <stdint.h>

typedef __attribute__((ext_vector_type(16))) _Float16 v16h;
typedef __attribute__((ext_vector_type(8)))  float    v8f;
typedef __attribute__((ext_vector_type(4)))  float    v4f;
typedef __attribute__((ext_vector_type(4)))  unsigned int v4u;

namespace {
constexpr int NF  = 30000;
constexpr int NM  = 3000;
constexpr int EFF = 180000;
constexpr int EMF = 60000;
constexpr int D   = 32;
constexpr int DM  = 64;
constexpr int AD  = 64;
constexpr int KW  = 128;
constexpr int EF  = 6;
constexpr int LL  = 1;   // only the last layer's y_flow is live in the reference

// workspace offsets (in floats)
constexpr size_t OFF_TEF   = 64;                                // te @ enc_f_W[4:] + b   (32)
constexpr size_t OFF_TEM   = 128;                               // te @ enc_m_W1[3:] + b1 (64)
constexpr size_t OFF_SCALE = 192;                               // (32)
constexpr size_t OFF_SHIFT = 256;                               // (32)
constexpr size_t OFF_XF    = 320;                               // x_flow   NF*32
constexpr size_t OFF_YM    = OFF_XF   + (size_t)NF*D;           // y_memb   NM*64
constexpr size_t OFF_KM    = OFF_YM   + (size_t)NM*DM;          // K_m      NM*64
constexpr size_t OFF_VM    = OFF_KM   + (size_t)NM*DM;          // V_m      NM*64
constexpr size_t OFF_QF    = OFF_VM   + (size_t)NM*DM;          // Q_f      NF*64
constexpr size_t OFF_SC    = OFF_QF   + (size_t)NF*AD;          // scores   EMF
constexpr size_t OFF_EV    = OFF_SC   + EMF;                    // exp vals EMF
constexpr size_t OFF_DEG   = OFF_EV   + EMF;                    // deg      NF   (zero-region start)
constexpr size_t OFF_SMAX  = OFF_DEG  + NF;                     // seg max (ordered uint) NF
constexpr size_t OFF_SSUM  = OFF_SMAX + NF;                     // seg sum  NF
constexpr size_t OFF_XCP   = OFF_SSUM + NF;                     // pre-Wo cross  NF*64
constexpr size_t OFF_AGGR  = OFF_XCP  + (size_t)NF*DM;          // msg aggregate NF*32
constexpr size_t OFF_XC    = OFF_AGGR + (size_t)NF*D;           // (zero-region end) x_cross NF*32
constexpr size_t OFF_YPX   = OFF_XC   + (size_t)NF*D;           // y_flow + x_flow NF*32
constexpr size_t OFF_K3T   = OFF_YPX  + (size_t)NF*D;           // kW3^T f16: 1024x128 halfs (65536 slots)
constexpr size_t OFF_K2T   = OFF_K3T  + 65536;                  // kW2^T f16: 128x128 halfs (8192 slots)
constexpr size_t ZERO_N    = OFF_XC - OFF_DEG;

constexpr int MSG_LDS = 262144 /*kW3T*/ + 32768 /*8x(16x128) f16 stage*/ + 4096 /*kb3*/;
}

// ---------- helpers ----------
__device__ __forceinline__ unsigned f2ord(float f) {
  unsigned u = __float_as_uint(f);
  return (u & 0x80000000u) ? ~u : (u | 0x80000000u);
}
__device__ __forceinline__ float ord2f(unsigned u) {
  return (u & 0x80000000u) ? __uint_as_float(u & 0x7fffffffu) : __uint_as_float(~u);
}

// assemble a 16-half WMMA fragment from two 16-byte segments
__device__ __forceinline__ v16h ld16_split(const _Float16* lo, const _Float16* hi) {
  v4u a = *(const v4u*)lo;
  v4u b = *(const v4u*)hi;
  v16h r;
  __builtin_memcpy(&r, &a, 16);
  __builtin_memcpy(reinterpret_cast<char*>(&r) + 16, &b, 16);
  return r;
}

// gfx1250 async DMA: 16 bytes per lane, global -> LDS, tracked by ASYNCcnt
__device__ __forceinline__ void async_g2l_b128(unsigned lds_off, const void* gaddr) {
  asm volatile("global_load_async_to_lds_b128 %0, %1, off"
               :: "v"(lds_off), "v"(gaddr) : "memory");
}

// ---------- tiny setup: timestep embedding, te-projections, per-layer scale/shift ----------
__global__ void setup_kernel(const float* __restrict__ tau_p,
                             const float* __restrict__ encfW, const float* __restrict__ encfb,
                             const float* __restrict__ encmW1, const float* __restrict__ encmb1,
                             const float* __restrict__ tW1l, const float* __restrict__ tb1l,
                             const float* __restrict__ tW2l, const float* __restrict__ tb2l,
                             float* __restrict__ ws) {
  __shared__ float te[32], th[64];
  int t = threadIdx.x;
  float tau = tau_p[0];
  if (t < 16) {
    float fr = expf(-logf(10000.f) * (float)t / 16.f);
    float a  = tau * fr;
    te[t] = sinf(a); te[16 + t] = cosf(a);
  }
  if (t < 64) { float z = tau * tW1l[t] + tb1l[t]; th[t] = z / (1.f + expf(-z)); }
  __syncthreads();
  if (t < 32) {
    float v = encfb[t];
    for (int j = 0; j < 32; ++j) v += te[j] * encfW[(4 + j) * 32 + t];
    ws[OFF_TEF + t] = v;
  }
  if (t < 64) {
    float v = encmb1[t];
    for (int j = 0; j < 32; ++j) v += te[j] * encmW1[(3 + j) * 64 + t];
    ws[OFF_TEM + t] = v;
    float s = tb2l[t];
    for (int k = 0; k < 64; ++k) s += th[k] * tW2l[k * 64 + t];
    if (t < 32) ws[OFF_SCALE + t] = s; else ws[OFF_SHIFT + (t - 32)] = s;
  }
}

// ---------- flow encoder: x_flow = [flow_x | te] @ enc_f_W + b ----------
__global__ void encode_flow_kernel(const float* __restrict__ fx, const float* __restrict__ W,
                                   const float* __restrict__ tef, float* __restrict__ xfo) {
  int t = blockIdx.x * blockDim.x + threadIdx.x;
  if (t >= NF * 32) return;
  int n = t >> 5, d = t & 31;
  float v = tef[d];
  #pragma unroll
  for (int i = 0; i < 4; ++i) v += fx[n * 4 + i] * W[i * 32 + d];
  xfo[t] = v;
}

// ---------- membrane encoder: two relu MLP layers ----------
__global__ void encode_memb_kernel(const float* __restrict__ my, const float* __restrict__ W1,
                                   const float* __restrict__ W2, const float* __restrict__ b2,
                                   const float* __restrict__ tem, float* __restrict__ ym) {
  __shared__ float hsm[64];
  int n = blockIdx.x, t = threadIdx.x;
  float v = tem[t];
  for (int i = 0; i < 3; ++i) v += my[n * 3 + i] * W1[i * 64 + t];
  hsm[t] = fmaxf(v, 0.f);
  __syncthreads();
  float v2 = b2[t];
  for (int k = 0; k < 64; ++k) v2 += hsm[k] * W2[k * 64 + t];
  ym[n * 64 + t] = fmaxf(v2, 0.f);
}

// ---------- K_m = y_memb@Wk + bk ; V_m = y_memb@Wv + bv (per membrane node) ----------
__global__ void kmvm_kernel(const float* __restrict__ ym, const float* __restrict__ Wk,
                            const float* __restrict__ bk, const float* __restrict__ Wv,
                            const float* __restrict__ bv, float* __restrict__ km,
                            float* __restrict__ vm) {
  __shared__ float ysm[64];
  int n = blockIdx.x, t = threadIdx.x;
  ysm[t] = ym[n * 64 + t];
  __syncthreads();
  float kv = bk[t], vv = bv[t];
  for (int j = 0; j < 64; ++j) { float y = ysm[j]; kv += y * Wk[j * 64 + t]; vv += y * Wv[j * 64 + t]; }
  km[n * 64 + t] = kv; vm[n * 64 + t] = vv;
}

// ---------- Q_f = x_flow @ Wq + bq ----------
__global__ void qf_kernel(const float* __restrict__ xf, const float* __restrict__ Wq,
                          const float* __restrict__ bq, float* __restrict__ qf) {
  int t = blockIdx.x * blockDim.x + threadIdx.x;
  if (t >= NF * 64) return;
  int n = t >> 6, d = t & 63;
  float v = bq[d];
  for (int j = 0; j < 32; ++j) v += xf[n * 32 + j] * Wq[j * 64 + d];
  qf[t] = v;
}

// ---------- degree of flow-flow destination nodes ----------
__global__ void deg_kernel(const int* __restrict__ ei, float* __restrict__ deg) {
  int e = blockIdx.x * blockDim.x + threadIdx.x;
  if (e < EFF) atomicAdd(&deg[ei[EFF + e]], 1.f);
}

// ---------- cross-attention: scores + ordered-uint atomic segment max ----------
__global__ void score_kernel(const int* __restrict__ srcm, const int* __restrict__ dstm,
                             const float* __restrict__ eam, const float* __restrict__ km,
                             const float* __restrict__ qf, const float* __restrict__ Weg,
                             const float* __restrict__ beg, float* __restrict__ score,
                             unsigned* __restrict__ smax) {
  int e = blockIdx.x * blockDim.x + threadIdx.x;
  if (e >= EMF) return;
  int s = srcm[e], dd = dstm[e];
  float ea[6];
  #pragma unroll
  for (int i = 0; i < 6; ++i) ea[i] = eam[e * 6 + i];
  float sc = 0.f;
  for (int d = 0; d < 64; ++d) {
    float kv = km[s * 64 + d] + beg[d];
    #pragma unroll
    for (int i = 0; i < 6; ++i) kv += ea[i] * Weg[i * 64 + d];
    sc += qf[(size_t)dd * 64 + d] * kv;
  }
  sc *= 0.125f;                       // 1/sqrt(64)
  score[e] = sc;
  atomicMax(&smax[dd], f2ord(sc));
}

__global__ void evsum_kernel(const int* __restrict__ dstm, const float* __restrict__ score,
                             const unsigned* __restrict__ smax, float* __restrict__ ev,
                             float* __restrict__ ssum) {
  int e = blockIdx.x * blockDim.x + threadIdx.x;
  if (e >= EMF) return;
  int dd = dstm[e];
  float x = expf(score[e] - ord2f(smax[dd]));
  ev[e] = x;
  atomicAdd(&ssum[dd], x);
}

__global__ void xcp_kernel(const int* __restrict__ srcm, const int* __restrict__ dstm,
                           const float* __restrict__ ev, const float* __restrict__ ssum,
                           const float* __restrict__ vm, float* __restrict__ xcp) {
  int t = blockIdx.x * blockDim.x + threadIdx.x;
  if (t >= EMF * 64) return;
  int e = t >> 6, d = t & 63;
  int dd = dstm[e];
  float alpha = ev[e] / ssum[dd];
  atomicAdd(&xcp[(size_t)dd * 64 + d], alpha * vm[(size_t)srcm[e] * 64 + d]);
}

__global__ void xc_kernel(const float* __restrict__ xcp, const float* __restrict__ Wo,
                          const float* __restrict__ bo, float* __restrict__ xc) {
  int t = blockIdx.x * blockDim.x + threadIdx.x;
  if (t >= NF * 32) return;
  int n = t >> 5, d = t & 31;
  float v = bo[d];
  for (int k = 0; k < 64; ++k) v += xcp[(size_t)n * 64 + k] * Wo[k * 32 + d];
  xc[t] = v;
}

// ---------- transpose + f16-cast kW3 (128x1024 -> 1024x128) and kW2 (128x128) ----------
__global__ void transpose_weights(const float* __restrict__ kW3g, const float* __restrict__ kW2g,
                                  _Float16* __restrict__ k3t, _Float16* __restrict__ k2t) {
  int t = blockIdx.x * blockDim.x + threadIdx.x;
  if (t < 131072) {
    int j = t >> 7, k = t & 127;
    k3t[t] = (_Float16)kW3g[k * 1024 + j];
  } else if (t < 131072 + 16384) {
    int u = t - 131072;
    int ko = u >> 7, ki = u & 127;
    k2t[u] = (_Float16)kW2g[ki * 128 + ko];
  }
}

// ---------- THE BIG ONE: fused edge hypernetwork + message + aggregation ----------
// per-wave 16-edge tiles; WMMA f32<-f16 for he2 = relu(he1@kW2) and W_edge = he2@kW3;
// kW3^T DMA'd into LDS (256 KB of the WGP's 320 KB) via global_load_async_to_lds_b128;
// msg contraction in VALU; atomic scatter by destination node.
__global__ __launch_bounds__(256)
void msg_kernel(const int* __restrict__ ei, const float* __restrict__ eattr,
                const float* __restrict__ kW1g, const float* __restrict__ kb1g,
                const float* __restrict__ kb2g, const float* __restrict__ kb3g,
                const _Float16* __restrict__ kW2T, const _Float16* __restrict__ kW3T,
                const float* __restrict__ xf, float* __restrict__ aggr) {
  extern __shared__ char smraw[];
  _Float16* lds_k3t   = (_Float16*)smraw;                    // 262144 B : kW3^T [1024][128] f16
  _Float16* lds_stage = (_Float16*)(smraw + 262144);         // 32768 B  : per-wave 16x128 f16
  float*    lds_kb3   = (float*)(smraw + 262144 + 32768);    // 4096 B

  const int tid = threadIdx.x;
  {
    // stage kW3^T into LDS with async DMA loads (16 B per lane per op, ASYNCcnt-tracked)
    const unsigned lds_base = (unsigned)(uintptr_t)(void*)lds_k3t;
    const char*    gbase    = (const char*)kW3T;
    for (int k = 0; k < 64; ++k) {
      unsigned off = (unsigned)((k * 256 + tid) * 16);       // 64 * 256 * 16 B = 256 KB
      async_g2l_b128(lds_base + off, gbase + off);
    }
    for (int i = tid; i < 1024; i += 256) lds_kb3[i] = kb3g[i];
    asm volatile("s_wait_asynccnt 0" ::: "memory");
  }
  __syncthreads();

  const int wid  = tid >> 5;
  const int lane = tid & 31;
  const int h    = lane >> 4;   // half-wave select
  const int e16  = lane & 15;   // edge-in-tile / N index
  _Float16* stage = lds_stage + wid * 2048;

  // loop-invariant per-lane weight columns (lane owns k = lane + 32q of he1)
  float w1c[4][6], b1c[4], b2c[8];
  #pragma unroll
  for (int q = 0; q < 4; ++q) {
    b1c[q] = kb1g[lane + 32 * q];
    #pragma unroll
    for (int i = 0; i < 6; ++i) w1c[q][i] = kW1g[i * 128 + lane + 32 * q];
  }
  #pragma unroll
  for (int ch = 0; ch < 8; ++ch) b2c[ch] = kb2g[ch * 16 + e16];

  const int nTiles = EFF / 16;
  const int wstep  = gridDim.x * 8;
  for (int tile = blockIdx.x * 8 + wid; tile < nTiles; tile += wstep) {
    const int base = tile * 16;

    // prefetch next tile's gather stream (lowers to global_prefetch_b8)
    const int nt = tile + wstep;
    if (nt < nTiles) {
      __builtin_prefetch(&eattr[(size_t)nt * 16 * 6], 0, 0);
      __builtin_prefetch(&ei[nt * 16], 0, 0);
      __builtin_prefetch(&ei[EFF + nt * 16], 0, 0);
    }

    // ---- he1 = relu(eattr @ kW1 + kb1) into stage[16][128] (f16, row-major)
    float ea[6];
    #pragma unroll
    for (int i = 0; i < 6; ++i) ea[i] = eattr[(size_t)(base + e16) * 6 + i];
    #pragma unroll
    for (int e = 0; e < 16; ++e) {
      float b0 = __shfl(ea[0], e, 32), b1 = __shfl(ea[1], e, 32), b2 = __shfl(ea[2], e, 32);
      float b3 = __shfl(ea[3], e, 32), b4 = __shfl(ea[4], e, 32), b5 = __shfl(ea[5], e, 32);
      #pragma unroll
      for (int q = 0; q < 4; ++q) {
        float v = b1c[q] + b0 * w1c[q][0] + b1 * w1c[q][1] + b2 * w1c[q][2]
                         + b3 * w1c[q][3] + b4 * w1c[q][4] + b5 * w1c[q][5];
        stage[e * 128 + lane + 32 * q] = (_Float16)fmaxf(v, 0.f);
      }
    }
    asm volatile("s_wait_dscnt 0" ::: "memory");   // cross-lane stage handoff

    // ---- he1 A-fragments (M=edge 16 x K=128): split-K layout per ISA table
    v16h afr[4];
    #pragma unroll
    for (int kb = 0; kb < 4; ++kb)
      afr[kb] = ld16_split(stage + e16 * 128 + kb * 32 + 8 * h,
                           stage + e16 * 128 + kb * 32 + 16 + 8 * h);

    // ---- he2 = relu(he1 @ kW2 + kb2) via WMMA; write back into stage
    #pragma unroll
    for (int ch = 0; ch < 8; ++ch) {
      v8f acc = (v8f)(0.0f);
      #pragma unroll
      for (int kb = 0; kb < 4; ++kb) {
        const _Float16* bp = kW2T + (size_t)(ch * 16 + e16) * 128 + kb * 32 + 8 * h;
        v16h bfr = ld16_split(bp, bp + 16);
        acc = __builtin_amdgcn_wmma_f32_16x16x32_f16(false, afr[kb], false, bfr,
                                                     (short)0, acc, false, false);
      }
      float bias = b2c[ch];
      #pragma unroll
      for (int p = 0; p < 8; ++p)
        stage[(p + 8 * h) * 128 + ch * 16 + e16] = (_Float16)fmaxf(acc[p] + bias, 0.f);
    }
    asm volatile("s_wait_dscnt 0" ::: "memory");

    // ---- B fragments = he2^T (K=128 x N=16 edges), split-K layout per ISA table
    v16h bfr[4];
    #pragma unroll
    for (int kb = 0; kb < 4; ++kb) {
      const _Float16* bp = stage + e16 * 128 + kb * 32 + 8 * h;
      bfr[kb] = ld16_split(bp, bp + 16);
    }

    // ---- x_src halves for this lane: xr[p]=x[8h+p], xr[8+p]=x[16+8h+p]
    const int s = ei[base + e16];
    const float* xrow = xf + (size_t)s * 32;
    float xr[16];
    {
      v4f a = *(const v4f*)(xrow + 8 * h);
      v4f b = *(const v4f*)(xrow + 8 * h + 4);
      v4f c = *(const v4f*)(xrow + 16 + 8 * h);
      v4f d = *(const v4f*)(xrow + 16 + 8 * h + 4);
      #pragma unroll
      for (int p = 0; p < 4; ++p) { xr[p] = a[p]; xr[4 + p] = b[p]; xr[8 + p] = c[p]; xr[12 + p] = d[p]; }
    }

    float msgr[32];
    #pragma unroll
    for (int i = 0; i < 32; ++i) msgr[i] = 0.f;

    // ---- 64 chunks of 16 W_edge columns: D[m=col,n=edge] = kW3^T-chunk @ he2^T
    #pragma unroll
    for (int c = 0; c < 64; ++c) {
      const _Float16* jrow = lds_k3t + (size_t)(c * 16 + e16) * 128;
      v8f acc = (v8f)(0.0f);
      #pragma unroll
      for (int kb = 0; kb < 4; ++kb) {
        v16h af = ld16_split(jrow + kb * 32 + 8 * h, jrow + kb * 32 + 16 + 8 * h);
        acc = __builtin_amdgcn_wmma_f32_16x16x32_f16(false, af, false, bfr[kb],
                                                     (short)0, acc, false, false);
      }
      // flat col = 16c + p + 8h  ->  i = c>>1,  j = (c&1)*16 + p + 8h
      float partial = 0.f;
      #pragma unroll
      for (int p = 0; p < 8; ++p) {
        float w = acc[p] + lds_kb3[c * 16 + p + 8 * h];
        partial += w * xr[(c & 1) * 8 + p];
      }
      float full = partial + __shfl_xor(partial, 16, 32);
      msgr[c >> 1] += full;
    }

    // ---- scatter to destination nodes (lanes 0..15 own the 16 edges)
    const int dn = ei[EFF + base + e16];
    if (h == 0) {
      #pragma unroll
      for (int i = 0; i < 32; ++i)
        atomicAdd(&aggr[(size_t)dn * 32 + i], msgr[i]);
    }
  }
}

// ---------- combine: x_intra + x_cross, FiLM, silu, residual ----------
__global__ void combine_kernel(const float* __restrict__ xf, const float* __restrict__ Wroot,
                               const float* __restrict__ broot, const float* __restrict__ aggr,
                               const float* __restrict__ deg, const float* __restrict__ xc,
                               const float* __restrict__ scale, const float* __restrict__ shift,
                               float* __restrict__ ypx) {
  int t = blockIdx.x * blockDim.x + threadIdx.x;
  if (t >= NF * 32) return;
  int n = t >> 5, d = t & 31;
  float v = broot[d] + aggr[t] / fmaxf(deg[n], 1.f);
  for (int j = 0; j < 32; ++j) v += xf[n * 32 + j] * Wroot[j * 32 + d];
  float g = fmaxf(v + xc[t], 0.f);
  float z = g * (1.f + scale[d]) + shift[d];
  float y = z / (1.f + expf(-z));
  ypx[t] = y + xf[t];
}

// ---------- decoders ----------
__global__ void decf_kernel(const float* __restrict__ ypx, const float* __restrict__ W,
                            const float* __restrict__ b, float* __restrict__ out) {
  int t = blockIdx.x * blockDim.x + threadIdx.x;
  if (t >= NF * 4) return;
  int n = t >> 2, o = t & 3;
  float v = b[o];
  for (int d = 0; d < 32; ++d) v += ypx[n * 32 + d] * W[d * 4 + o];
  out[t] = v;
}

__global__ void decm_kernel(const float* __restrict__ ym, const float* __restrict__ W,
                            const float* __restrict__ b, float* __restrict__ out) {
  int t = blockIdx.x * blockDim.x + threadIdx.x;
  if (t >= NM * 3) return;
  int n = t / 3, o = t % 3;
  float v = b[o];
  for (int d = 0; d < 64; ++d) v += ym[n * 64 + d] * W[d * 3 + o];
  out[t] = v;
}

extern "C" void kernel_launch(void* const* d_in, const int* in_sizes, int n_in,
                              void* d_out, int out_size, void* d_ws, size_t ws_size,
                              hipStream_t stream) {
  (void)in_sizes; (void)n_in; (void)out_size; (void)ws_size;

  const float* flow_x   = (const float*)d_in[0];
  const float* memb_y   = (const float*)d_in[1];
  const float* tau      = (const float*)d_in[2];
  const int*   ei_ff    = (const int*)d_in[3];
  const float* eattr_ff = (const float*)d_in[4];
  const int*   mf_src   = (const int*)d_in[5];
  const int*   mf_dst   = (const int*)d_in[6];
  const float* eattr_mf = (const float*)d_in[7];
  const float* enc_f_W  = (const float*)d_in[8];
  const float* enc_f_b  = (const float*)d_in[9];
  const float* enc_m_W1 = (const float*)d_in[10];
  const float* enc_m_b1 = (const float*)d_in[11];
  const float* enc_m_W2 = (const float*)d_in[12];
  const float* enc_m_b2 = (const float*)d_in[13];
  // per-layer weights, offset to the only live layer LL=1
  const float* kW1   = (const float*)d_in[14] + (size_t)LL * EF * KW;
  const float* kb1   = (const float*)d_in[15] + (size_t)LL * KW;
  const float* kW2   = (const float*)d_in[16] + (size_t)LL * KW * KW;
  const float* kb2   = (const float*)d_in[17] + (size_t)LL * KW;
  const float* kW3   = (const float*)d_in[18] + (size_t)LL * KW * D * D;
  const float* kb3   = (const float*)d_in[19] + (size_t)LL * D * D;
  const float* Wroot = (const float*)d_in[20] + (size_t)LL * D * D;
  const float* broot = (const float*)d_in[21] + (size_t)LL * D;
  const float* Wq    = (const float*)d_in[22] + (size_t)LL * D * AD;
  const float* bq    = (const float*)d_in[23] + (size_t)LL * AD;
  const float* Wk    = (const float*)d_in[24] + (size_t)LL * DM * AD;
  const float* bk    = (const float*)d_in[25] + (size_t)LL * AD;
  const float* We    = (const float*)d_in[26] + (size_t)LL * EF * AD;
  const float* be    = (const float*)d_in[27] + (size_t)LL * AD;
  const float* Wv    = (const float*)d_in[28] + (size_t)LL * DM * AD;
  const float* bv    = (const float*)d_in[29] + (size_t)LL * AD;
  const float* Wo    = (const float*)d_in[30] + (size_t)LL * AD * D;
  const float* bo    = (const float*)d_in[31] + (size_t)LL * D;
  const float* tW1   = (const float*)d_in[32] + (size_t)LL * DM;
  const float* tb1   = (const float*)d_in[33] + (size_t)LL * DM;
  const float* tW2   = (const float*)d_in[34] + (size_t)LL * DM * 2 * D;
  const float* tb2   = (const float*)d_in[35] + (size_t)LL * 2 * D;
  const float* dec_f_W = (const float*)d_in[36];
  const float* dec_f_b = (const float*)d_in[37];
  const float* dec_m_W = (const float*)d_in[38];
  const float* dec_m_b = (const float*)d_in[39];

  float* ws   = (float*)d_ws;
  float* xf   = ws + OFF_XF;
  float* ym   = ws + OFF_YM;
  float* km   = ws + OFF_KM;
  float* vm   = ws + OFF_VM;
  float* qf   = ws + OFF_QF;
  float* score= ws + OFF_SC;
  float* ev   = ws + OFF_EV;
  float* deg  = ws + OFF_DEG;
  unsigned* smax = (unsigned*)(ws + OFF_SMAX);
  float* ssum = ws + OFF_SSUM;
  float* xcp  = ws + OFF_XCP;
  float* aggr = ws + OFF_AGGR;
  float* xc   = ws + OFF_XC;
  float* ypx  = ws + OFF_YPX;
  _Float16* k3t = (_Float16*)(ws + OFF_K3T);
  _Float16* k2t = (_Float16*)(ws + OFF_K2T);

  float* outF = (float*)d_out;
  float* outM = outF + (size_t)NF * 4;

  hipMemsetAsync(ws + OFF_DEG, 0, ZERO_N * sizeof(float), stream);

  setup_kernel<<<1, 64, 0, stream>>>(tau, enc_f_W, enc_f_b, enc_m_W1, enc_m_b1,
                                     tW1, tb1, tW2, tb2, ws);
  encode_flow_kernel<<<(NF * 32 + 255) / 256, 256, 0, stream>>>(flow_x, enc_f_W, ws + OFF_TEF, xf);
  encode_memb_kernel<<<NM, 64, 0, stream>>>(memb_y, enc_m_W1, enc_m_W2, enc_m_b2, ws + OFF_TEM, ym);
  kmvm_kernel<<<NM, 64, 0, stream>>>(ym, Wk, bk, Wv, bv, km, vm);
  qf_kernel<<<(NF * 64 + 255) / 256, 256, 0, stream>>>(xf, Wq, bq, qf);
  deg_kernel<<<(EFF + 255) / 256, 256, 0, stream>>>(ei_ff, deg);
  transpose_weights<<<(131072 + 16384 + 255) / 256, 256, 0, stream>>>(kW3, kW2, k3t, k2t);
  score_kernel<<<(EMF + 255) / 256, 256, 0, stream>>>(mf_src, mf_dst, eattr_mf, km, qf, We, be,
                                                      score, smax);
  evsum_kernel<<<(EMF + 255) / 256, 256, 0, stream>>>(mf_dst, score, smax, ev, ssum);
  xcp_kernel<<<(EMF * 64 + 255) / 256, 256, 0, stream>>>(mf_src, mf_dst, ev, ssum, vm, xcp);
  xc_kernel<<<(NF * 32 + 255) / 256, 256, 0, stream>>>(xcp, Wo, bo, xc);
  msg_kernel<<<256, 256, MSG_LDS, stream>>>(ei_ff, eattr_ff, kW1, kb1, kb2, kb3, k2t, k3t, xf, aggr);
  combine_kernel<<<(NF * 32 + 255) / 256, 256, 0, stream>>>(xf, Wroot, broot, aggr, deg, xc,
                                                            ws + OFF_SCALE, ws + OFF_SHIFT, ypx);
  decf_kernel<<<(NF * 4 + 255) / 256, 256, 0, stream>>>(ypx, dec_f_W, dec_f_b, outF);
  decm_kernel<<<(NM * 3 + 255) / 256, 256, 0, stream>>>(ym, dec_m_W, dec_m_b, outM);
}
